// MultiScaleDeformableAttention_35923106463881
// MI455X (gfx1250) — compile-verified, matrix operand order checked by reading, not run on hardware
//
#include <hip/hip_runtime.h>

// ---------------------------------------------------------------------------
// Multi-scale deformable attention for MI455X (gfx1250, wave32, WMMA).
// Pipeline: 3x WMMA GEMM projections -> fused softmax + bilinear gather ->
// WMMA GEMM output projection.
// ---------------------------------------------------------------------------

typedef __attribute__((ext_vector_type(16))) __bf16    v16bf;
typedef __attribute__((ext_vector_type(8)))  float     v8f;
typedef __attribute__((ext_vector_type(4)))  float     fx4;
typedef __attribute__((ext_vector_type(4)))  unsigned  ux4;
typedef __attribute__((ext_vector_type(2)))  unsigned  ux2;

#define BM   128   // block tile M
#define BN   64    // block tile N
#define KT   64    // K staged per LDS round
#define KDIM 256   // embed dim (fixed by reference)
#define LDT  72    // padded LDS row stride in halves (144B -> conflict-free b128 reads)

static __device__ __forceinline__ unsigned pack_bf16x2(float lo, float hi) {
  union { float f; unsigned u; } a, b;
  a.f = lo; b.f = hi;
  unsigned ra = a.u + 0x7FFFu + ((a.u >> 16) & 1u);   // round-to-nearest-even
  unsigned rb = b.u + 0x7FFFu + ((b.u >> 16) & 1u);
  return (ra >> 16) | (rb & 0xFFFF0000u);
}

// C[M,N] = A[M,K] * W[N,K]^T + bias[N]   (K = 256, N multiple of 64)
__global__ __launch_bounds__(256) void gemm_bias_bf16wmma(
    const float* __restrict__ A,
    const float* __restrict__ W,
    const float* __restrict__ bias,
    float* __restrict__ C,
    int M, int N)
{
  __shared__ __align__(16) unsigned short sA[BM * LDT];
  __shared__ __align__(16) unsigned short sB[BN * LDT];

  const int tid  = threadIdx.x;
  const int lane = tid & 31;
  const int wave = tid >> 5;       // 0..7
  const int wm   = wave >> 1;      // 0..3  (M direction, 32 rows each)
  const int wn   = wave & 1;       // 0..1  (N direction, 32 cols each)
  const int m0   = blockIdx.x * BM;
  const int n0   = blockIdx.y * BN;

  const int r  = lane & 15;           // row/col within 16x16 fragment
  const int kb = (lane >> 4) << 3;    // K phase: lanes 0-15 -> 0, 16-31 -> 8

  // Bias folded into accumulator init: in the C/D layout every VGPR of a lane
  // maps to the same output column (n = n0 + wn*32 + j*16 + r), so a splat of
  // bias[col] is an exact initialization.
  v8f acc[2][2];
  {
    const float bv0 = bias[n0 + wn * 32 + r];
    const float bv1 = bias[n0 + wn * 32 + 16 + r];
    union { v8f v; float f[8]; } s0, s1;
    #pragma unroll
    for (int e = 0; e < 8; ++e) { s0.f[e] = bv0; s1.f[e] = bv1; }
    acc[0][0] = s0.v; acc[1][0] = s0.v;
    acc[0][1] = s1.v; acc[1][1] = s1.v;
  }

  // Staging coordinates (loop-invariant)
  const int arow = tid >> 1;
  const int acb  = (tid & 1) << 5;               // 0 or 32
  int agr = m0 + arow; if (agr > M - 1) agr = M - 1;   // clamp: EXEC stays full
  const float* aptr = A + (size_t)agr * KDIM + acb;
  const int brow = tid >> 2;
  const int bcb  = (tid & 3) << 4;               // 0,16,32,48
  const float* bptr = W + (size_t)(n0 + brow) * KDIM + bcb;

  for (int ks = 0; ks < KDIM; ks += KT) {
    // ---- stage A tile (128 x 64 f32 -> bf16), 2 threads per row ----
    {
      const fx4* gp = (const fx4*)(aptr + ks);
      unsigned short* dp = &sA[arow * LDT + acb];
      if (ks + KT < KDIM) __builtin_prefetch(aptr + ks + KT);   // global_prefetch_b8
      #pragma unroll
      for (int i = 0; i < 8; ++i) {
        fx4 f = gp[i];
        ux2 p; p.x = pack_bf16x2(f.x, f.y); p.y = pack_bf16x2(f.z, f.w);
        *(ux2*)(dp + i * 4) = p;
      }
    }
    // ---- stage B tile (64 x 64 f32 -> bf16), 4 threads per row ----
    {
      const fx4* gp = (const fx4*)(bptr + ks);
      unsigned short* dp = &sB[brow * LDT + bcb];
      #pragma unroll
      for (int i = 0; i < 4; ++i) {
        fx4 f = gp[i];
        ux2 p; p.x = pack_bf16x2(f.x, f.y); p.y = pack_bf16x2(f.z, f.w);
        *(ux2*)(dp + i * 4) = p;
      }
    }
    __syncthreads();

    // ---- 2 WMMA K-steps of 32 over the staged 64 ----
    #pragma unroll
    for (int kk = 0; kk < KT; kk += 32) {
      union Frag { ux4 q[2]; v16bf bf; } a0, a1, b0, b1;
      const unsigned short* pa0 = &sA[(wm * 32 + r) * LDT + kk + kb];
      const unsigned short* pa1 = pa0 + 16 * LDT;
      const unsigned short* pb0 = &sB[(wn * 32 + r) * LDT + kk + kb];
      const unsigned short* pb1 = pb0 + 16 * LDT;
      a0.q[0] = *(const ux4*)pa0;        a0.q[1] = *(const ux4*)(pa0 + 16);
      a1.q[0] = *(const ux4*)pa1;        a1.q[1] = *(const ux4*)(pa1 + 16);
      b0.q[0] = *(const ux4*)pb0;        b0.q[1] = *(const ux4*)(pb0 + 16);
      b1.q[0] = *(const ux4*)pb1;        b1.q[1] = *(const ux4*)(pb1 + 16);

      acc[0][0] = __builtin_amdgcn_wmma_f32_16x16x32_bf16(false, a0.bf, false, b0.bf, (short)0, acc[0][0], false, false);
      acc[0][1] = __builtin_amdgcn_wmma_f32_16x16x32_bf16(false, a0.bf, false, b1.bf, (short)0, acc[0][1], false, false);
      acc[1][0] = __builtin_amdgcn_wmma_f32_16x16x32_bf16(false, a1.bf, false, b0.bf, (short)0, acc[1][0], false, false);
      acc[1][1] = __builtin_amdgcn_wmma_f32_16x16x32_bf16(false, a1.bf, false, b1.bf, (short)0, acc[1][1], false, false);
    }
    __syncthreads();
  }

  // ---- epilogue: store (bias already folded into acc) ----
  const int rbh  = (lane >> 4) << 3;
  const bool full = (m0 + BM) <= M;      // wave-uniform: interior block fast path
  #pragma unroll
  for (int j = 0; j < 2; ++j) {
    const int cn = n0 + wn * 32 + j * 16 + r;
    #pragma unroll
    for (int i = 0; i < 2; ++i) {
      const int mb = m0 + wm * 32 + i * 16 + rbh;
      union { v8f v; float f[8]; } u; u.v = acc[i][j];
      float* cp = C + (size_t)mb * N + cn;
      if (full) {
        #pragma unroll
        for (int e = 0; e < 8; ++e) { *cp = u.f[e]; cp += N; }
      } else {
        #pragma unroll
        for (int e = 0; e < 8; ++e) { if (mb + e < M) *cp = u.f[e]; cp += N; }
      }
    }
  }
}

// ---------------------------------------------------------------------------
// Fused softmax + multi-scale bilinear gather.
// One wave per (batch*query, head); 32 lanes = 32 head channels, so each
// corner gather is a coalesced 128B read of the projected value row.
// vproj (~41MB) is L2-resident (192MB L2), so corner gathers never hit HBM.
// ---------------------------------------------------------------------------
#define LVLS 4
#define NPTS 4
#define HEADS 8

__global__ __launch_bounds__(256) void msda_softmax_kernel(
    const float* __restrict__ vproj,   // (bs*nv, 256) = (b, pos, head, 32)
    const float* __restrict__ offs,    // (bs*nq, 256) = h*L*p*2
    const float* __restrict__ awraw,   // (bs*nq, 128) = h*L*p
    const float* __restrict__ refp,    // (bs*nq, L, 2)
    const int*   __restrict__ shapes,  // (L, 2) = (H, W)
    float* __restrict__ out,           // (bs*nq, 256)
    int Mq)                            // bs * nq
{
  const int gwave = (int)((blockIdx.x * blockDim.x + threadIdx.x) >> 5);
  const int lane  = threadIdx.x & 31;
  const int bq    = gwave >> 3;        // / HEADS
  const int head  = gwave & (HEADS - 1);
  if (bq >= Mq) return;

  int Hs[LVLS], Ws[LVLS], starts[LVLS];
  int nv = 0;
  #pragma unroll
  for (int l = 0; l < LVLS; ++l) {
    Hs[l] = shapes[2 * l];
    Ws[l] = shapes[2 * l + 1];
    starts[l] = nv;
    nv += Hs[l] * Ws[l];
  }
  const int b = bq / nv;               // nq == nv in this problem
  const size_t vbase = (size_t)b * nv * (HEADS * 32) + head * 32 + lane;

  // softmax over the 16 (level, point) logits of this head
  float w[LVLS * NPTS];
  const float* ap = awraw + (size_t)bq * (HEADS * LVLS * NPTS) + head * (LVLS * NPTS);
  float mx = -1e30f;
  #pragma unroll
  for (int i = 0; i < LVLS * NPTS; ++i) { w[i] = ap[i]; mx = fmaxf(mx, w[i]); }
  float s = 0.f;
  #pragma unroll
  for (int i = 0; i < LVLS * NPTS; ++i) { w[i] = __expf(w[i] - mx); s += w[i]; }
  const float inv = 1.0f / s;

  const float* op = offs + (size_t)bq * 256 + head * (LVLS * NPTS * 2);
  const float* rp = refp + (size_t)bq * (LVLS * 2);

  float acc = 0.f;
  #pragma unroll
  for (int l = 0; l < LVLS; ++l) {
    const float Wf = (float)Ws[l], Hf = (float)Hs[l];
    const int   Wi = Ws[l],        Hi = Hs[l];
    const float rx = rp[2 * l], ry = rp[2 * l + 1];
    const float* vl = vproj + vbase + (size_t)starts[l] * (HEADS * 32);
    #pragma unroll
    for (int p = 0; p < NPTS; ++p) {
      const float ox = op[(l * NPTS + p) * 2];
      const float oy = op[(l * NPTS + p) * 2 + 1];
      // loc = ref + off/[W,H]; x = loc_x*W - 0.5 = rx*W + ox - 0.5
      const float x = rx * Wf + ox - 0.5f;
      const float y = ry * Hf + oy - 0.5f;
      const float x0f = floorf(x), y0f = floorf(y);
      const float wx = x - x0f, wy = y - y0f;
      const int x0 = (int)x0f, y0 = (int)y0f;
      const float aww = w[l * NPTS + p] * inv;
      float sample = 0.f;
      {
        int xx = x0, yy = y0;
        if (xx >= 0 && xx < Wi && yy >= 0 && yy < Hi)
          sample += (1.f - wx) * (1.f - wy) * vl[(size_t)(yy * Wi + xx) * (HEADS * 32)];
      }
      {
        int xx = x0 + 1, yy = y0;
        if (xx >= 0 && xx < Wi && yy >= 0 && yy < Hi)
          sample += wx * (1.f - wy) * vl[(size_t)(yy * Wi + xx) * (HEADS * 32)];
      }
      {
        int xx = x0, yy = y0 + 1;
        if (xx >= 0 && xx < Wi && yy >= 0 && yy < Hi)
          sample += (1.f - wx) * wy * vl[(size_t)(yy * Wi + xx) * (HEADS * 32)];
      }
      {
        int xx = x0 + 1, yy = y0 + 1;
        if (xx >= 0 && xx < Wi && yy >= 0 && yy < Hi)
          sample += wx * wy * vl[(size_t)(yy * Wi + xx) * (HEADS * 32)];
      }
      acc += aww * sample;
    }
  }
  out[(size_t)bq * 256 + head * 32 + lane] = acc;
}

// ---------------------------------------------------------------------------

extern "C" void kernel_launch(void* const* d_in, const int* in_sizes, int n_in,
                              void* d_out, int out_size, void* d_ws, size_t ws_size,
                              hipStream_t stream) {
  const float* query  = (const float*)d_in[0];
  const float* value  = (const float*)d_in[1];
  const float* refpts = (const float*)d_in[2];
  const int*   shapes = (const int*)  d_in[3];
  const float* W_off  = (const float*)d_in[4];
  const float* b_off  = (const float*)d_in[5];
  const float* W_attn = (const float*)d_in[6];
  const float* b_attn = (const float*)d_in[7];
  const float* W_val  = (const float*)d_in[8];
  const float* b_val  = (const float*)d_in[9];
  const float* W_out  = (const float*)d_in[10];
  const float* b_out  = (const float*)d_in[11];
  float* out = (float*)d_out;

  const int K     = KDIM;                 // 256
  const int Mq    = in_sizes[0] / K;      // bs * nq
  const int Mv    = in_sizes[1] / K;      // bs * nv
  const int Nattn = in_sizes[6] / K;      // 128

  // workspace layout (f32): off | aw | vproj | msda
  float* ws_off = (float*)d_ws;
  float* ws_aw  = ws_off + (size_t)Mq * 256;
  float* ws_v   = ws_aw  + (size_t)Mq * Nattn;
  float* ws_ms  = ws_v   + (size_t)Mv * 256;

  const dim3 blk(256);
  const int gxq = (Mq + BM - 1) / BM;
  const int gxv = (Mv + BM - 1) / BM;

  // value projection
  gemm_bias_bf16wmma<<<dim3(gxv, 256 / BN), blk, 0, stream>>>(value, W_val, b_val, ws_v, Mv, 256);
  // sampling offsets
  gemm_bias_bf16wmma<<<dim3(gxq, 256 / BN), blk, 0, stream>>>(query, W_off, b_off, ws_off, Mq, 256);
  // attention logits
  gemm_bias_bf16wmma<<<dim3(gxq, Nattn / BN), blk, 0, stream>>>(query, W_attn, b_attn, ws_aw, Mq, Nattn);
  // fused softmax + deformable gather: Mq blocks x 8 waves = one wave per (bq, head)
  msda_softmax_kernel<<<dim3(Mq), blk, 0, stream>>>(ws_v, ws_off, ws_aw, refpts, shapes, ws_ms, Mq);
  // output projection
  gemm_bias_bf16wmma<<<dim3(gxq, 256 / BN), blk, 0, stream>>>(ws_ms, W_out, b_out, out, Mq, 256);
}
